// BitFlipLinearLora_80607946211595
// MI455X (gfx1250) — matrix operand order repaired and checked by reading, not compiled, not run on hardware
//
#include <hip/hip_runtime.h>

// ---------------------------------------------------------------------------
// BitFlipLinearLora for MI455X (gfx1250, wave32, WMMA, async-to-LDS staging).
//   Y[8192,4096] = bitflip(x) @ bitflip(W + lora_B@lora_A * scaling)^T
// Pipeline:
//   1) bitflip_x:       x fp32 -> bitflip -> bf16   (d_ws, 64 MB)
//   2) merge_bitflip_w: W + B@A -> bitflip -> bf16  (d_ws+64MB, 32 MB)
//   3) wmma_gemm:       bf16 x bf16 -> fp32, v_wmma_f32_16x16x32_bf16,
//                       GLOBAL_LOAD_ASYNC_TO_LDS_B128, triple-buffered with
//                       two stages of DMA in flight (s_wait_asynccnt 4).
// Workspace requirement: ~100.7 MB.
// ---------------------------------------------------------------------------

typedef unsigned int   u32;
typedef unsigned short u16;
typedef unsigned long long u64;
typedef __attribute__((ext_vector_type(4)))  u32    u32x4;
typedef __attribute__((ext_vector_type(2)))  u32    u32x2;
typedef __attribute__((ext_vector_type(4)))  float  f32x4;
typedef __attribute__((ext_vector_type(8)))  float  f32x8;
typedef __attribute__((ext_vector_type(16))) __bf16 bf16x16;

union Frag { bf16x16 v; u32x4 q[2]; };

// problem config (mirrors reference)
constexpr int GM = 8192;   // B*S
constexpr int GN = 4096;   // D_OUT
constexpr int GK = 4096;   // D_IN
constexpr int LR = 32;     // LoRA rank; SCALING = 32/32 = 1.0
// seeds: _BASE = 0 -> x_exp=0, x_frac=1, w_exp=2, w_frac=3

// ------------------------- bitflip PRNG helpers ----------------------------
__device__ __forceinline__ u32 mixu(u32 x) {
  x ^= x >> 16; x *= 0x7feb352dU;
  x ^= x >> 15; x *= 0x846ca68bU;
  x ^= x >> 16;
  return x;
}

// AND of HALVES uniform u32 draws -> each bit set with prob 2^-HALVES
template <int HALVES>
__device__ __forceinline__ u32 flip_mask(u32 seed, u32 idx, u32 region) {
  u32 m = 0xFFFFFFFFu;
  u32 s = mixu(idx * 0x9E3779B9u ^ (seed * 0x85EBCA77u + 0x165667B1u));
#pragma unroll
  for (int i = 0; i < HALVES; ++i) {
    s = s * 0x2C9277B5u + 0xAC564B05u;  // LCG advance per draw
    m &= mixu(s);
  }
  return m & region;
}

template <int EH, int FH>
__device__ __forceinline__ float bitflip(float v, u32 idx, u32 se, u32 sf, float zt) {
  u32 b = __float_as_uint(v);
  b ^= flip_mask<EH>(se, idx, 0x7F800000u);  // exponent bits
  b ^= flip_mask<FH>(sf, idx, 0x007FFFFFu);  // fraction bits
  float y = __uint_as_float(b);
  bool finite = (b & 0x7F800000u) != 0x7F800000u;
  return (finite && fabsf(y) <= zt) ? y : 0.0f;
}

__device__ __forceinline__ u16 to_bf16(float f) {  // round-to-nearest-even
  u32 u = __float_as_uint(f);
  u += 0x7FFFu + ((u >> 16) & 1u);
  return (u16)(u >> 16);
}

// ------------------------- kernel 1: bitflip(x) -> bf16 --------------------
__global__ __launch_bounds__(256) void bitflip_x_kernel(const float* __restrict__ x,
                                                        u16* __restrict__ xb) {
  size_t e = ((size_t)blockIdx.x * 256 + threadIdx.x) * 4;
  f32x4 v = *(const f32x4*)(x + e);
  u32 idx = (u32)e;
  u16 h0 = to_bf16(bitflip<10, 6>(v.x, idx + 0, 0u, 1u, 1000.0f));
  u16 h1 = to_bf16(bitflip<10, 6>(v.y, idx + 1, 0u, 1u, 1000.0f));
  u16 h2 = to_bf16(bitflip<10, 6>(v.z, idx + 2, 0u, 1u, 1000.0f));
  u16 h3 = to_bf16(bitflip<10, 6>(v.w, idx + 3, 0u, 1u, 1000.0f));
  u32x2 p;
  p.x = (u32)h0 | ((u32)h1 << 16);
  p.y = (u32)h2 | ((u32)h3 << 16);
  *(u32x2*)(xb + e) = p;
}

// --------- kernel 2: bitflip(W + lora_B@lora_A) -> bf16 (64x64 tiles) ------
__global__ __launch_bounds__(256) void merge_bitflip_w_kernel(
    const float* __restrict__ w, const float* __restrict__ lA,
    const float* __restrict__ lB, u16* __restrict__ wb) {
  __shared__ float sB[64 * LR];   // lora_B rows for this o-tile  (8 KB)
  __shared__ float sA[LR * 64];   // lora_A cols for this i-tile  (8 KB)
  const int tid = threadIdx.x;
  const int bO = blockIdx.y * 64;  // output-feature tile
  const int bI = blockIdx.x * 64;  // input-feature tile

  // lora_B[bO..bO+63][0..31] is contiguous (row-major [4096][32])
  {
    const f32x4* src = (const f32x4*)(lB + (size_t)bO * LR);
    f32x4* dst = (f32x4*)sB;
    dst[tid]       = src[tid];
    dst[tid + 256] = src[tid + 256];
  }
  // lora_A[r][bI..bI+63], 32 rows x 16 float4-chunks
  {
#pragma unroll
    for (int j = 0; j < 2; ++j) {
      int c = tid + j * 256;
      int r = c >> 4, cc = (c & 15) * 4;
      *(f32x4*)(sA + r * 64 + cc) = *(const f32x4*)(lA + (size_t)r * GK + bI + cc);
    }
  }
  __syncthreads();

  const int to = tid >> 4;   // 4 output rows per thread
  const int ti = tid & 15;   // 4 input cols per thread
  float acc[4][4] = {};
#pragma unroll
  for (int r = 0; r < LR; ++r) {
    float a0 = sA[r * 64 + ti * 4 + 0];
    float a1 = sA[r * 64 + ti * 4 + 1];
    float a2 = sA[r * 64 + ti * 4 + 2];
    float a3 = sA[r * 64 + ti * 4 + 3];
#pragma unroll
    for (int q = 0; q < 4; ++q) {
      float bb = sB[(to * 4 + q) * LR + r];
      acc[q][0] += bb * a0;
      acc[q][1] += bb * a1;
      acc[q][2] += bb * a2;
      acc[q][3] += bb * a3;
    }
  }

#pragma unroll
  for (int q = 0; q < 4; ++q) {
    int o = bO + to * 4 + q;
    int i = bI + ti * 4;
    f32x4 wv = *(const f32x4*)(w + (size_t)o * GK + i);
    u32 base = (u32)o * (u32)GK + (u32)i;
    // SCALING == 1.0f
    u16 h0 = to_bf16(bitflip<12, 8>(wv.x + acc[q][0], base + 0, 2u, 3u, 100.0f));
    u16 h1 = to_bf16(bitflip<12, 8>(wv.y + acc[q][1], base + 1, 2u, 3u, 100.0f));
    u16 h2 = to_bf16(bitflip<12, 8>(wv.z + acc[q][2], base + 2, 2u, 3u, 100.0f));
    u16 h3 = to_bf16(bitflip<12, 8>(wv.w + acc[q][3], base + 3, 2u, 3u, 100.0f));
    u32x2 p;
    p.x = (u32)h0 | ((u32)h1 << 16);
    p.y = (u32)h2 | ((u32)h3 << 16);
    *(u32x2*)(wb + (size_t)o * GK + i) = p;
  }
}

// ------------------- kernel 3: bf16 WMMA GEMM (NT layout) ------------------
// C[m,n] = sum_k A[m,k]*B[n,k]; A,B row-major bf16 (K-major), C fp32.
// Block: 256 threads = 8 waves, tile 128x128, K-stage 32, triple-buffered LDS
// filled with GLOBAL_LOAD_ASYNC_TO_LDS_B128 (ASYNCcnt-tracked DMA), keeping
// two stages in flight: s_wait_asynccnt 4 relies on in-order async completion.
// Wave grid 2(M)x4(N); wave tile 64x32 -> 4x2 accumulator fragments.
constexpr int BM = 128, BN = 128, BK = 32;
constexpr int LDT = 40;                 // padded LDS row stride (bf16 elems)
constexpr int TILE_ELEMS = BM * LDT;    // 5120
constexpr int BUF_ELEMS = 2 * TILE_ELEMS;
constexpr int NBUF = 3;                 // 3 x 20 KB = 60 KB LDS

// 16-byte async DMA: global -> LDS, tracked by ASYNCcnt.
// VDST VGPR carries the LDS byte address (flat-pointer low 32 bits == LDS offset).
__device__ __forceinline__ void async_ld16(const u16* lds_ptr, const u16* gptr) {
  u32 lds_addr = (u32)(size_t)lds_ptr;
  u64 gaddr = (u64)(size_t)gptr;
  asm volatile("global_load_async_to_lds_b128 %0, %1, off"
               :
               : "v"(lds_addr), "v"(gaddr)
               : "memory");
}
template <int N>
__device__ __forceinline__ void wait_async() {
  asm volatile("s_wait_asynccnt %0" :: "n"(N) : "memory");
}

__global__ __launch_bounds__(256) void wmma_gemm_kernel(const u16* __restrict__ A,
                                                        const u16* __restrict__ B,
                                                        float* __restrict__ C) {
  __shared__ __align__(16) u16 sm[NBUF * BUF_ELEMS];  // 60 KB

  const int tid = threadIdx.x;
  const int wave = tid >> 5, lane = tid & 31;
  const int wm = wave & 1;    // wave M index (0..1)
  const int wn = wave >> 1;   // wave N index (0..3)
  const int lm = lane & 15, lh = lane >> 4;
  const int bM = blockIdx.y * BM, bN = blockIdx.x * BN;

  // staging geometry: each thread DMAs 2x16B per tile (A and B) = 4 loads/stage
  const int r0 = tid >> 2;            // rows 0..63
  const int c0 = (tid & 3) * 8;       // 8 bf16 = 16 B chunks across BK=32

  auto async_stage = [&](int kt, int buf) {
    u16* pa = sm + buf * BUF_ELEMS;
    u16* pb = pa + TILE_ELEMS;
    const u16* agp = A + (size_t)(bM + r0) * GK + kt * BK + c0;
    const u16* bgp = B + (size_t)(bN + r0) * GK + kt * BK + c0;
    async_ld16(pa + r0 * LDT + c0, agp);
    async_ld16(pa + (r0 + 64) * LDT + c0, agp + (size_t)64 * GK);
    async_ld16(pb + r0 * LDT + c0, bgp);
    async_ld16(pb + (r0 + 64) * LDT + c0, bgp + (size_t)64 * GK);
  };

  f32x8 acc[4][2];
  const f32x8 zero = {0.f, 0.f, 0.f, 0.f, 0.f, 0.f, 0.f, 0.f};
#pragma unroll
  for (int mi = 0; mi < 4; ++mi)
#pragma unroll
    for (int ni = 0; ni < 2; ++ni) acc[mi][ni] = zero;

  // prologue: two stages in flight; stage 0 must be resident before compute
  async_stage(0, 0);
  async_stage(1, 1);
  wait_async<4>();   // 8 outstanding -> <=4 means stage 0 landed (in-order)
  __syncthreads();

  const int NKT = GK / BK;  // 128
  for (int kt = 0; kt < NKT; ++kt) {
    const int buf = kt % NBUF;
    // kick off DMA two stages ahead into the retired buffer
    if (kt + 2 < NKT) async_stage(kt + 2, (kt + 2) % NBUF);

    const u16* pa = sm + buf * BUF_ELEMS;
    const u16* pb = pa + TILE_ELEMS;

    // A fragment (16x32 bf16): lane<16 holds K 0..7 & 16..23, lane>=16 K 8..15 & 24..31
    Frag af[4];
#pragma unroll
    for (int mi = 0; mi < 4; ++mi) {
      const u16* p = pa + (wm * 64 + mi * 16 + lm) * LDT + lh * 8;
      af[mi].q[0] = *(const u32x4*)p;
      af[mi].q[1] = *(const u32x4*)(p + 16);
    }
    // B fragment (32x16 bf16, N = lane%16): lane<16 K 0..7 & 16..23, lane>=16 K 8..15 & 24..31
    Frag bfr[2];
#pragma unroll
    for (int ni = 0; ni < 2; ++ni) {
      const u16* p = pb + (wn * 32 + ni * 16 + lm) * LDT + lh * 8;
      bfr[ni].q[0] = *(const u32x4*)p;
      bfr[ni].q[1] = *(const u32x4*)(p + 16);
    }

#pragma unroll
    for (int mi = 0; mi < 4; ++mi)
#pragma unroll
      for (int ni = 0; ni < 2; ++ni)
        acc[mi][ni] = __builtin_amdgcn_wmma_f32_16x16x32_bf16(
            false, af[mi].v, false, bfr[ni].v, (short)0, acc[mi][ni], false, false);

    // ensure stage kt+1 (older 4 loads) landed; stage kt+2 stays in flight.
    if (kt + 2 < NKT) wait_async<4>();
    else              wait_async<0>();
    __syncthreads();
  }

  // epilogue: C/D layout — VGPR r: lanes 0-15 M=r, lanes 16-31 M=8+r; N=lane%16
#pragma unroll
  for (int mi = 0; mi < 4; ++mi) {
#pragma unroll
    for (int ni = 0; ni < 2; ++ni) {
      int row0 = bM + wm * 64 + mi * 16 + lh * 8;
      int col = bN + wn * 32 + ni * 16 + lm;
#pragma unroll
      for (int r = 0; r < 8; ++r)
        C[(size_t)(row0 + r) * GN + col] = acc[mi][ni][r];
    }
  }
}

// ---------------------------------------------------------------------------
extern "C" void kernel_launch(void* const* d_in, const int* in_sizes, int n_in,
                              void* d_out, int out_size, void* d_ws, size_t ws_size,
                              hipStream_t stream) {
  const float* x      = (const float*)d_in[0];  // [4,2048,4096]
  const float* weight = (const float*)d_in[1];  // [4096,4096]
  const float* lora_A = (const float*)d_in[2];  // [32,4096]
  const float* lora_B = (const float*)d_in[3];  // [4096,32]
  float* out = (float*)d_out;                   // [4,2048,4096]

  u16* xb = (u16*)d_ws;                         // 8192*4096 bf16 = 64 MB
  u16* wb = xb + (size_t)GM * GK;               // 4096*4096 bf16 = 32 MB

  bitflip_x_kernel<<<(GM * GK) / (256 * 4), 256, 0, stream>>>(x, xb);
  merge_bitflip_w_kernel<<<dim3(GK / 64, GN / 64), 256, 0, stream>>>(weight, lora_A, lora_B, wb);
  wmma_gemm_kernel<<<dim3(GN / BN, GM / BM), 256, 0, stream>>>(xb, wb, out);
}
